// PointNetFeaturePropagation_30056181138039
// MI455X (gfx1250) — matrix operand order repaired and checked by reading, not compiled, not run on hardware
//
#include <hip/hip_runtime.h>

// ---------------------------------------------------------------------------
// Problem constants (match reference)
// ---------------------------------------------------------------------------
#define BB   8
#define NN   8192
#define SS   2048
#define D1C  128
#define D2C  256
#define C0   384      // IN_CH = D1 + D2
#define C1   256      // MLP[0]
#define C2   128      // MLP[1]
#define MTOT (BB * NN)  // 65536 rows
#define EPSV 1e-5f

// ---------------------------------------------------------------------------
// WMMA types (CDNA5, wave32)
// ---------------------------------------------------------------------------
typedef __attribute__((ext_vector_type(16))) __bf16 v16bf;
typedef __attribute__((ext_vector_type(8)))  float  v8f;

union FragBF16 {
    uint4 u[2];     // 32 bytes
    v16bf v;        // 16 x bf16 = 32 bytes
};

// float -> bf16 (round-to-nearest-even)
__device__ __forceinline__ unsigned short f2bf(float f) {
    unsigned u = __float_as_uint(f);
    unsigned r = (u >> 16) & 1u;
    u += 0x7fffu + r;
    return (unsigned short)(u >> 16);
}

// ---------------------------------------------------------------------------
// Small utility kernels
// ---------------------------------------------------------------------------
__global__ void zero_f32_kernel(float* __restrict__ p, int n) {
    int i = blockIdx.x * blockDim.x + threadIdx.x;
    if (i < n) p[i] = 0.0f;
}

__global__ void f32_to_bf16_kernel(const float* __restrict__ in,
                                   unsigned short* __restrict__ out, int n) {
    int i = blockIdx.x * blockDim.x + threadIdx.x;
    if (i < n) out[i] = f2bf(in[i]);
}

// ---------------------------------------------------------------------------
// Kernel 1: 3-NN search + inverse-distance interpolation + concat -> bf16 X
// One wave32 per query point. X layout: [MTOT][384] bf16 row-major.
// ---------------------------------------------------------------------------
__global__ void knn_interp_concat_kernel(const float* __restrict__ xyz1,
                                         const float* __restrict__ xyz2,
                                         const float* __restrict__ points1,
                                         const float* __restrict__ points2,
                                         unsigned short* __restrict__ X) {
    const int gw   = (blockIdx.x * blockDim.x + threadIdx.x) >> 5; // point id
    const int lane = threadIdx.x & 31;
    if (gw >= MTOT) return;
    const int b = gw / NN;

    const float* q = xyz1 + (size_t)gw * 3;
    const float qx = q[0], qy = q[1], qz = q[2];

    // per-lane local top-3 over 64 candidates (s = lane, lane+32, ...)
    float d0 = 3.4e38f, d1 = 3.4e38f, d2 = 3.4e38f;
    int   i0 = -1, i1 = -1, i2 = -1;
    const float* base2 = xyz2 + (size_t)b * SS * 3;
    for (int s = lane; s < SS; s += 32) {
        const float* r = base2 + (size_t)s * 3;
        float dx = qx - r[0], dy = qy - r[1], dz = qz - r[2];
        float d = dx * dx + dy * dy + dz * dz;
        if (d < d2) {
            if (d < d0)      { d2 = d1; i2 = i1; d1 = d0; i1 = i0; d0 = d; i0 = s; }
            else if (d < d1) { d2 = d1; i2 = i1; d1 = d;  i1 = s; }
            else             { d2 = d;  i2 = s; }
        }
    }

    // 3 rounds of wave-wide arg-min (butterfly leaves result in all lanes)
    float wd[3]; int wi[3];
    #pragma unroll
    for (int k = 0; k < 3; ++k) {
        float cd = d0; int ci = i0;
        #pragma unroll
        for (int off = 16; off > 0; off >>= 1) {
            float od = __shfl_xor(cd, off);
            int   oi = __shfl_xor(ci, off);
            if (od < cd || (od == cd && oi < ci)) { cd = od; ci = oi; }
        }
        wd[k] = cd; wi[k] = ci;
        if (ci == i0) {  // only owner lane pops (indices unique across lanes)
            d0 = d1; i0 = i1; d1 = d2; i1 = i2; d2 = 3.4e38f; i2 = -1;
        }
    }

    // inverse-distance weights
    float w[3]; float sw = 0.0f;
    #pragma unroll
    for (int k = 0; k < 3; ++k) {
        float dk = wd[k] < 1e-10f ? 1e-10f : wd[k];
        w[k] = 1.0f / dk; sw += w[k];
    }
    const float invs = 1.0f / sw;
    #pragma unroll
    for (int k = 0; k < 3; ++k) w[k] *= invs;

    unsigned short* row = X + (size_t)gw * C0;

    // copy points1: 128 channels, 4 per lane -> bf16
    {
        const float4 p = *(const float4*)(points1 + (size_t)gw * D1C + lane * 4);
        uint2 o;
        o.x = (unsigned)f2bf(p.x) | ((unsigned)f2bf(p.y) << 16);
        o.y = (unsigned)f2bf(p.z) | ((unsigned)f2bf(p.w) << 16);
        *(uint2*)(row + lane * 4) = o;
    }

    // interpolate points2: 256 channels, 8 per lane
    {
        float acc[8];
        #pragma unroll
        for (int r = 0; r < 8; ++r) acc[r] = 0.0f;
        #pragma unroll
        for (int k = 0; k < 3; ++k) {
            const float* src = points2 + ((size_t)b * SS + wi[k]) * D2C + lane * 8;
            const float4 a = *(const float4*)(src);
            const float4 c = *(const float4*)(src + 4);
            acc[0] += w[k] * a.x; acc[1] += w[k] * a.y;
            acc[2] += w[k] * a.z; acc[3] += w[k] * a.w;
            acc[4] += w[k] * c.x; acc[5] += w[k] * c.y;
            acc[6] += w[k] * c.z; acc[7] += w[k] * c.w;
        }
        uint4 o;
        o.x = (unsigned)f2bf(acc[0]) | ((unsigned)f2bf(acc[1]) << 16);
        o.y = (unsigned)f2bf(acc[2]) | ((unsigned)f2bf(acc[3]) << 16);
        o.z = (unsigned)f2bf(acc[4]) | ((unsigned)f2bf(acc[5]) << 16);
        o.w = (unsigned)f2bf(acc[6]) | ((unsigned)f2bf(acc[7]) << 16);
        *(uint4*)(row + D1C + lane * 8) = o;
    }
}

// ---------------------------------------------------------------------------
// Kernel 2: bf16 GEMM via v_wmma_f32_16x16x32_bf16, 32x32 register-blocked.
//   Y[M,N] = X[M,K] @ W[N,K]^T + bias
// One wave32 computes a 32x32 output tile: 2 A fragments x 2 B fragments
// -> 4 WMMAs per K-step from 8 b128 loads (2 loads/WMMA vs 4 unblocked).
// Fragment layouts follow ISA 7.12.2:
//   A (16x32 bf16): lane l -> row m0+(l&15); elems 0..7 : K = k0 + (l>>4)*8 + e
//                                            elems 8..15: K = k0 + 16 + (l>>4)*8 + (e-8)
//   B (32x16 bf16): lane l -> col n0+(l&15); same K mapping as A
//   C/D (16x16 f32): lane l -> col n0+(l&15); vgpr r    : M = m0 + (l>>4)*8 + r
// Fused: bias preloaded into accumulators; per-channel BN sum/sumsq epilogue.
// ---------------------------------------------------------------------------
__global__ void gemm_bias_stats_kernel(const unsigned short* __restrict__ Xb,
                                       const unsigned short* __restrict__ Wb,
                                       const float* __restrict__ bias,
                                       float* __restrict__ Y,
                                       float* __restrict__ sum,
                                       float* __restrict__ sumsq,
                                       int M, int N, int K) {
    const int gw   = (blockIdx.x * blockDim.x + threadIdx.x) >> 5;
    const int lane = threadIdx.x & 31;
    const int ntiles = N >> 5;               // 32-wide N tiles
    const int tm = gw / ntiles;
    const int tn = gw - tm * ntiles;
    if ((tm << 5) >= M) return;

    const int half = lane >> 4;
    const int mr   = lane & 15;
    const int m0   = tm << 5;
    const int n0   = tn << 5;
    const int nn0  = n0 + mr;                // channel for ni=0
    const int nn1  = n0 + 16 + mr;           // channel for ni=1

    const float bv0 = bias[nn0];
    const float bv1 = bias[nn1];
    v8f acc00, acc01, acc10, acc11;          // acc[mi][ni]
    #pragma unroll
    for (int r = 0; r < 8; ++r) {
        acc00[r] = bv0; acc01[r] = bv1;
        acc10[r] = bv0; acc11[r] = bv1;
    }

    const unsigned short* arow0 = Xb + (size_t)(m0 + mr) * K;
    const unsigned short* arow1 = Xb + (size_t)(m0 + 16 + mr) * K;
    const unsigned short* brow0 = Wb + (size_t)nn0 * K;
    const unsigned short* brow1 = Wb + (size_t)nn1 * K;

    for (int k0 = 0; k0 < K; k0 += 32) {
        FragBF16 A0, A1, B0, B1;
        A0.u[0] = *(const uint4*)(arow0 + k0 + half * 8);
        A0.u[1] = *(const uint4*)(arow0 + k0 + 16 + half * 8);
        A1.u[0] = *(const uint4*)(arow1 + k0 + half * 8);
        A1.u[1] = *(const uint4*)(arow1 + k0 + 16 + half * 8);
        B0.u[0] = *(const uint4*)(brow0 + k0 + half * 8);
        B0.u[1] = *(const uint4*)(brow0 + k0 + 16 + half * 8);
        B1.u[0] = *(const uint4*)(brow1 + k0 + half * 8);
        B1.u[1] = *(const uint4*)(brow1 + k0 + 16 + half * 8);
        acc00 = __builtin_amdgcn_wmma_f32_16x16x32_bf16(
            false, A0.v, false, B0.v, (short)0, acc00, false, false);
        acc01 = __builtin_amdgcn_wmma_f32_16x16x32_bf16(
            false, A0.v, false, B1.v, (short)0, acc01, false, false);
        acc10 = __builtin_amdgcn_wmma_f32_16x16x32_bf16(
            false, A1.v, false, B0.v, (short)0, acc10, false, false);
        acc11 = __builtin_amdgcn_wmma_f32_16x16x32_bf16(
            false, A1.v, false, B1.v, (short)0, acc11, false, false);
    }

    // store fp32 results + per-channel partial BN statistics
    float s1n0 = 0.0f, s2n0 = 0.0f, s1n1 = 0.0f, s2n1 = 0.0f;
    #pragma unroll
    for (int r = 0; r < 8; ++r) {
        const int ma = m0 + half * 8 + r;        // rows of mi=0 tile
        const int mb = ma + 16;                  // rows of mi=1 tile
        const float v00 = acc00[r], v01 = acc01[r];
        const float v10 = acc10[r], v11 = acc11[r];
        Y[(size_t)ma * N + nn0] = v00;
        Y[(size_t)ma * N + nn1] = v01;
        Y[(size_t)mb * N + nn0] = v10;
        Y[(size_t)mb * N + nn1] = v11;
        s1n0 += v00 + v10; s2n0 += v00 * v00 + v10 * v10;
        s1n1 += v01 + v11; s2n1 += v01 * v01 + v11 * v11;
    }
    // lanes l and l+16 hold the same channel: fold then atomically accumulate
    s1n0 += __shfl_down(s1n0, 16);
    s2n0 += __shfl_down(s2n0, 16);
    s1n1 += __shfl_down(s1n1, 16);
    s2n1 += __shfl_down(s2n1, 16);
    if (lane < 16) {
        atomicAdd(&sum[nn0],   s1n0);
        atomicAdd(&sumsq[nn0], s2n0);
        atomicAdd(&sum[nn1],   s1n1);
        atomicAdd(&sumsq[nn1], s2n1);
    }
}

// ---------------------------------------------------------------------------
// Kernel 3: BatchNorm (training stats) + ReLU
// ---------------------------------------------------------------------------
__global__ void bn_relu_bf16_kernel(const float* __restrict__ Y,
                                    const float* __restrict__ sum,
                                    const float* __restrict__ sumsq,
                                    const float* __restrict__ gamma,
                                    const float* __restrict__ beta,
                                    unsigned short* __restrict__ Xout,
                                    int M, int N) {
    const size_t idx = (size_t)blockIdx.x * blockDim.x + threadIdx.x;
    if (idx >= (size_t)M * N) return;
    const int c = (int)(idx % (size_t)N);
    const float invM = 1.0f / (float)M;
    const float mean = sum[c] * invM;
    const float var  = sumsq[c] * invM - mean * mean;
    float v = (Y[idx] - mean) * rsqrtf(var + EPSV) * gamma[c] + beta[c];
    v = v > 0.0f ? v : 0.0f;
    Xout[idx] = f2bf(v);
}

__global__ void bn_relu_f32_kernel(const float* __restrict__ Y,
                                   const float* __restrict__ sum,
                                   const float* __restrict__ sumsq,
                                   const float* __restrict__ gamma,
                                   const float* __restrict__ beta,
                                   float* __restrict__ out,
                                   int M, int N) {
    const size_t idx = (size_t)blockIdx.x * blockDim.x + threadIdx.x;
    if (idx >= (size_t)M * N) return;
    const int c = (int)(idx % (size_t)N);
    const float invM = 1.0f / (float)M;
    const float mean = sum[c] * invM;
    const float var  = sumsq[c] * invM - mean * mean;
    float v = (Y[idx] - mean) * rsqrtf(var + EPSV) * gamma[c] + beta[c];
    out[idx] = v > 0.0f ? v : 0.0f;
}

// ---------------------------------------------------------------------------
// Launcher
// ---------------------------------------------------------------------------
extern "C" void kernel_launch(void* const* d_in, const int* in_sizes, int n_in,
                              void* d_out, int out_size, void* d_ws, size_t ws_size,
                              hipStream_t stream) {
    const float* xyz1    = (const float*)d_in[0];
    const float* xyz2    = (const float*)d_in[1];
    const float* points1 = (const float*)d_in[2];
    const float* points2 = (const float*)d_in[3];
    const float* w0      = (const float*)d_in[4];
    const float* b0      = (const float*)d_in[5];
    const float* gamma0  = (const float*)d_in[6];
    const float* beta0   = (const float*)d_in[7];
    const float* w1      = (const float*)d_in[8];
    const float* b1      = (const float*)d_in[9];
    const float* gamma1  = (const float*)d_in[10];
    const float* beta1   = (const float*)d_in[11];
    float* out = (float*)d_out;

    // workspace carve-up (256B aligned)
    char* ws = (char*)d_ws;
    size_t off = 0;
    auto carve = [&](size_t bytes) -> void* {
        void* p = ws + off;
        off = (off + bytes + 255) & ~(size_t)255;
        return p;
    };
    unsigned short* X   = (unsigned short*)carve((size_t)MTOT * C0 * 2); // bf16 concat
    unsigned short* X2  = (unsigned short*)carve((size_t)MTOT * C1 * 2); // bf16 layer-1 act
    float*          Y1  = (float*)carve((size_t)MTOT * C1 * 4);          // fp32 pre-BN
    float*          Y2  = Y1;   // alias: Y1 fully consumed before GEMM2 writes Y2
    unsigned short* W0b = (unsigned short*)carve((size_t)C1 * C0 * 2);
    unsigned short* W1b = (unsigned short*)carve((size_t)C2 * C1 * 2);
    float*          st  = (float*)carve(768 * 4);
    float* sum0 = st;          // [256]
    float* sq0  = st + 256;    // [256]
    float* sum1 = st + 512;    // [128]
    float* sq1  = st + 640;    // [128]

    // 0) zero BN accumulators (graph replays re-run this -> deterministic)
    zero_f32_kernel<<<3, 256, 0, stream>>>(st, 768);

    // 1) weights -> bf16
    f32_to_bf16_kernel<<<(C1 * C0 + 255) / 256, 256, 0, stream>>>(w0, W0b, C1 * C0);
    f32_to_bf16_kernel<<<(C2 * C1 + 255) / 256, 256, 0, stream>>>(w1, W1b, C2 * C1);

    // 2) 3-NN + interpolate + concat: one wave per point, 8 waves/block
    knn_interp_concat_kernel<<<MTOT / 8, 256, 0, stream>>>(
        xyz1, xyz2, points1, points2, X);

    // 3) layer 1 GEMM (65536x384 @ 384x256) + bias + stats (32x32 tiles)
    {
        const int waves = (MTOT / 32) * (C1 / 32);   // 16384 waves
        gemm_bias_stats_kernel<<<waves / 8, 256, 0, stream>>>(
            X, W0b, b0, Y1, sum0, sq0, MTOT, C1, C0);
    }

    // 4) BN + ReLU -> bf16 activations
    bn_relu_bf16_kernel<<<(int)(((size_t)MTOT * C1 + 255) / 256), 256, 0, stream>>>(
        Y1, sum0, sq0, gamma0, beta0, X2, MTOT, C1);

    // 5) layer 2 GEMM (65536x256 @ 256x128) + bias + stats (32x32 tiles)
    {
        const int waves = (MTOT / 32) * (C2 / 32);   // 8192 waves
        gemm_bias_stats_kernel<<<waves / 8, 256, 0, stream>>>(
            X2, W1b, b1, Y2, sum1, sq1, MTOT, C2, C1);
    }

    // 6) BN + ReLU -> fp32 output
    bn_relu_f32_kernel<<<(int)(((size_t)MTOT * C2 + 255) / 256), 256, 0, stream>>>(
        Y2, sum1, sq1, gamma1, beta1, out, MTOT, C2);
}
